// AttentionDecoderCell_12446815224573
// MI455X (gfx1250) — matrix operand order repaired and checked by reading, not compiled, not run on hardware
//
#include <hip/hip_runtime.h>
#include <hip/hip_bf16.h>

#define B_    32
#define T_    2048
#define D_    512
#define U_    256
#define TDEC_ 256

typedef __attribute__((ext_vector_type(16))) __bf16 v16bf;
typedef __attribute__((ext_vector_type(8)))  __bf16 v8bf;
typedef __attribute__((ext_vector_type(4)))  __bf16 v4bf;
typedef __attribute__((ext_vector_type(2)))  __bf16 v2bf;
typedef __attribute__((ext_vector_type(8)))  float  v8f;

#if __has_builtin(__builtin_amdgcn_tanhf)
static __device__ __forceinline__ float fast_tanh(float x) { return __builtin_amdgcn_tanhf(x); }
#else
static __device__ __forceinline__ float fast_tanh(float x) { return tanhf(x); }
#endif

static __device__ __forceinline__ float hs(float z) {
    return fminf(fmaxf(0.2f * z + 0.5f, 0.0f), 1.0f);
}

// ---- WMMA A fragment (16x32 bf16, MxK) from row-major bf16, contiguous K-pairs = 4B loads ----
static __device__ __forceinline__ v16bf load_a_bf16g(const __bf16* __restrict__ p, int lda, int k0) {
    int l = threadIdx.x & 31;
    int g = l >> 4, m = l & 15;
    v16bf a;
#pragma unroll
    for (int r = 0; r < 8; ++r) {
        int kb = (r < 4 ? 2 * r : 16 + 2 * (r - 4)) + g * 8;
        v2bf pv = *(const v2bf*)(p + (size_t)m * lda + k0 + kb);
        a[2 * r]     = pv[0];
        a[2 * r + 1] = pv[1];
    }
    return a;
}

// ---- WMMA A fragment from row-major f32 with on-the-fly bf16 convert (8B pair loads) ----
static __device__ __forceinline__ v16bf load_a_f32cvt(const float* __restrict__ p, int lda, int k0) {
    int l = threadIdx.x & 31;
    int g = l >> 4, m = l & 15;
    v16bf a;
#pragma unroll
    for (int r = 0; r < 8; ++r) {
        int kb = (r < 4 ? 2 * r : 16 + 2 * (r - 4)) + g * 8;
        const float* pr = p + (size_t)m * lda + k0 + kb;
        a[2 * r]     = (__bf16)pr[0];
        a[2 * r + 1] = (__bf16)pr[1];
    }
    return a;
}

// ---- Pre-packed B fragments: one contiguous 32B v16bf load per lane ----
static __device__ __forceinline__ v16bf load_b_packed(const __bf16* __restrict__ Bp, int Kdiv32,
                                                      int nTile, int kTile) {
    int l = threadIdx.x & 31;
    return *(const v16bf*)(Bp + (((size_t)nTile * Kdiv32 + kTile) * 32 + l) * 16);
}

// Pack f32 [K x N] row-major weights into WMMA-B fragment order (bf16).
// Fragment semantics: lane l (g=l>>4, n=l&15), elem e in [0,16): K = kTile*32 + (e < 8 ? g*8 + e : 16 + g*8 + (e - 8)), col = nTile*16 + n.
__global__ void pack_b(const float* __restrict__ W, __bf16* __restrict__ out, int K, int N) {
    int idx = blockIdx.x * blockDim.x + threadIdx.x;   // tile*32 + lane
    int lane = idx & 31;
    int tile = idx >> 5;
    int Kdiv32 = K >> 5;
    int nT = tile / Kdiv32;
    int kT = tile - nT * Kdiv32;
    if (nT >= (N >> 4)) return;
    int g = lane >> 4, n = lane & 15;
    __bf16* o = out + ((size_t)tile * 32 + lane) * 16;
#pragma unroll
    for (int e = 0; e < 16; ++e) {
        int k = kT * 32 + (e < 8 ? g * 8 + e : 16 + g * 8 + (e - 8));
        o[e] = (__bf16)W[(size_t)k * N + nT * 16 + n];
    }
}

// ---- x f32 -> bf16 (plain [b,t,d] layout), 4 elems/thread ----
__global__ void cvt_x(const float* __restrict__ in, __bf16* __restrict__ out, int n4) {
    int i = blockIdx.x * blockDim.x + threadIdx.x;
    if (i >= n4) return;
    float4 v = ((const float4*)in)[i];
    v4bf o;
    o[0] = (__bf16)v.x; o[1] = (__bf16)v.y; o[2] = (__bf16)v.z; o[3] = (__bf16)v.w;
    *(v4bf*)(out + 4 * (size_t)i) = o;
}

// ---- uxpb^T[b][u][t] = (x @ U_a + b_a)^T   (WMMA bf16; transposed store = 16B/lane) ----
__global__ void uxpb_gemm(const __bf16* __restrict__ xb, const __bf16* __restrict__ UaP,
                          const float* __restrict__ ba, __bf16* __restrict__ uxpbT) {
    int gw = blockIdx.x * (blockDim.x >> 5) + (threadIdx.x >> 5);
    int mTile = gw >> 4;   // 4096 tiles over B*T
    int nTile = gw & 15;   // 16 tiles over U
    int l = threadIdx.x & 31;
    int g = l >> 4, n = l & 15;
    float bias = ba[nTile * 16 + n];
    v8f acc;
#pragma unroll
    for (int r = 0; r < 8; ++r) acc[r] = bias;

    int row0 = mTile * 16;
    for (int k0 = 0; k0 < D_; k0 += 32) {
        v16bf a = load_a_bf16g(xb + (size_t)row0 * D_, D_, k0);
        v16bf b = load_b_packed(UaP, D_ / 32, nTile, k0 >> 5);
        acc = __builtin_amdgcn_wmma_f32_16x16x32_bf16(false, a, false, b, (short)0, acc, false, false);
    }
    // D rows m = r + 8g are consecutive t -> one contiguous v8bf store per lane
    int b    = row0 >> 11;         // / T_
    int tloc = row0 & (T_ - 1);
    v8bf st;
#pragma unroll
    for (int r = 0; r < 8; ++r) st[r] = (__bf16)acc[r];
    *(v8bf*)(uxpbT + ((size_t)(b * U_ + nTile * 16 + n)) * T_ + tloc + 8 * g) = st;
}

// ---- h0 = tanh(x[:,0,:] @ W_s), c0 = 0 ----
__global__ void init_state(const float* __restrict__ x, const float* __restrict__ Ws,
                           float* __restrict__ h, float* __restrict__ c) {
    int b = blockIdx.x, u = threadIdx.x;
    float acc = 0.f;
    const float* xr = x + (size_t)b * T_ * D_;
    for (int d = 0; d < D_; ++d) acc += xr[d] * Ws[d * U_ + u];
    h[b * U_ + u] = tanhf(acc);
    c[b * U_ + u] = 0.f;
}

// ---- e[b,t] = sum_u V_a[u]*tanh(uxpbT[b,u,t] + (h@W_a)[b,u]); also zeroes ctx ----
__global__ void attn_scores(const __bf16* __restrict__ uxpbT, const float* __restrict__ h,
                            const float* __restrict__ Wa, const float* __restrict__ Va,
                            float* __restrict__ e, float* __restrict__ ctx) {
    __shared__ float2 hv2[U_];
    int b = blockIdx.x >> 2;
    int tchunk = blockIdx.x & 3;
    int tid = threadIdx.x;

    if (blockIdx.x < (B_ * D_) / 256) ctx[blockIdx.x * 256 + tid] = 0.f;  // zero for next kernel's atomics

    // hWa[b, u=tid]: h broadcast, Wa coalesced
    float acc = 0.f;
    const float* hb = h + b * U_;
    for (int k = 0; k < U_; ++k) acc += hb[k] * Wa[k * U_ + tid];
    hv2[tid] = make_float2(acc, Va[tid]);
    __syncthreads();

    int t = tchunk * 512 + tid * 2;                       // 2 consecutive t per thread
    const __bf16* col = uxpbT + t;
    float s0 = 0.f, s1 = 0.f;
    for (int u = 0; u < U_; ++u) {
        float2 hv = hv2[u];                               // single 8B LDS broadcast
        v2bf v = *(const v2bf*)(col + (size_t)(b * U_ + u) * T_);  // 4B/lane, contiguous per wave
        s0 += hv.y * fast_tanh((float)v[0] + hv.x);
        s1 += hv.y * fast_tanh((float)v[1] + hv.x);
    }
    e[b * T_ + t]     = s0;
    e[b * T_ + t + 1] = s1;
}

// ---- softmax over t (redundant per block) + partial ctx over a 256-t slice; atomic combine ----
__global__ void softmax_ctx(const float* __restrict__ e, const __bf16* __restrict__ xb,
                            float* __restrict__ ctx) {
    __shared__ float red[256];
    __shared__ float a_sh[256];
    int b = blockIdx.x >> 3;
    int tchunk = blockIdx.x & 7;
    int tid = threadIdx.x;

    const float* er = e + b * T_;
    float lv[8];
    float mx = -3.4e38f;
#pragma unroll
    for (int i = 0; i < 8; ++i) { lv[i] = er[tid + i * 256]; mx = fmaxf(mx, lv[i]); }
    red[tid] = mx; __syncthreads();
    for (int s = 128; s > 0; s >>= 1) { if (tid < s) red[tid] = fmaxf(red[tid], red[tid + s]); __syncthreads(); }
    mx = red[0]; __syncthreads();

    float ls = 0.f;
#pragma unroll
    for (int i = 0; i < 8; ++i) { lv[i] = __expf(lv[i] - mx); ls += lv[i]; }
    red[tid] = ls; __syncthreads();
    for (int s = 128; s > 0; s >>= 1) { if (tid < s) red[tid] += red[tid + s]; __syncthreads(); }
    float inv = 1.f / red[0];

    a_sh[tid] = lv[tchunk] * inv;    // this block's t-slice: t = tchunk*256 + tid
    __syncthreads();

    int d0 = tid * 2;                // all 512 d per block, 2/thread
    const __bf16* xr = xb + (size_t)(b * T_ + tchunk * 256) * D_ + d0;
    float a0 = 0.f, a1 = 0.f;
    for (int t = 0; t < 256; ++t) {
        float w = a_sh[t];
        v2bf v = *(const v2bf*)(xr + (size_t)t * D_);   // 4B/lane, 1KB contiguous per block-row
        a0 += w * (float)v[0];
        a1 += w * (float)v[1];
    }
    atomicAdd(&ctx[b * D_ + d0],     a0);
    atomicAdd(&ctx[b * D_ + d0 + 1], a1);
}

// ---- z = ctx@kernel + h@recurrent + bias; fused LSTM pointwise + state update ----
__global__ void lstm_gates(const float* __restrict__ ctx, const __bf16* __restrict__ kernP,
                           const float* __restrict__ h_in, const __bf16* __restrict__ recP,
                           const float* __restrict__ bias,
                           float* __restrict__ h_out, float* __restrict__ c,
                           float* __restrict__ out, int step) {
    int w = blockIdx.x;       // 32 tiles: 2 M-tiles x 16 N-tiles
    int mTile = w >> 4;
    int nTile = w & 15;
    int l = threadIdx.x & 31;
    int g = l >> 4, n = l & 15;

    v8f acc[4];
#pragma unroll
    for (int gg = 0; gg < 4; ++gg) {
        float bv = bias[gg * U_ + nTile * 16 + n];
#pragma unroll
        for (int r = 0; r < 8; ++r) acc[gg][r] = bv;
    }

    // ctx @ kernel : K = 512; B packed per (gate-column tile, kTile)
    for (int k0 = 0; k0 < D_; k0 += 32) {
        v16bf a = load_a_f32cvt(ctx + (size_t)mTile * 16 * D_, D_, k0);
#pragma unroll
        for (int gg = 0; gg < 4; ++gg) {
            v16bf bb = load_b_packed(kernP, D_ / 32, gg * 16 + nTile, k0 >> 5);
            acc[gg] = __builtin_amdgcn_wmma_f32_16x16x32_bf16(false, a, false, bb, (short)0, acc[gg], false, false);
        }
    }
    // h @ recurrent : K = 256
    for (int k0 = 0; k0 < U_; k0 += 32) {
        v16bf a = load_a_f32cvt(h_in + (size_t)mTile * 16 * U_, U_, k0);
#pragma unroll
        for (int gg = 0; gg < 4; ++gg) {
            v16bf bb = load_b_packed(recP, U_ / 32, gg * 16 + nTile, k0 >> 5);
            acc[gg] = __builtin_amdgcn_wmma_f32_16x16x32_bf16(false, a, false, bb, (short)0, acc[gg], false, false);
        }
    }

#pragma unroll
    for (int r = 0; r < 8; ++r) {
        int m = r + 8 * g;
        int bidx = mTile * 16 + m;
        int u = nTile * 16 + n;
        float zi = acc[0][r], zf = acc[1][r], zc = acc[2][r], zo = acc[3][r];
        float iv = hs(zi), fv = hs(zf), ov = hs(zo);
        float cold = c[bidx * U_ + u];
        float cnew = fv * cold + iv * fast_tanh(zc);
        float hnew = ov * fast_tanh(cnew);
        c[bidx * U_ + u]     = cnew;
        h_out[bidx * U_ + u] = hnew;
        out[((size_t)bidx * TDEC_ + step) * U_ + u] = hnew;
    }
}

extern "C" void kernel_launch(void* const* d_in, const int* in_sizes, int n_in,
                              void* d_out, int out_size, void* d_ws, size_t ws_size,
                              hipStream_t stream) {
    const float* x    = (const float*)d_in[0];
    const float* Ws   = (const float*)d_in[1];
    const float* Ua   = (const float*)d_in[2];
    const float* ba   = (const float*)d_in[3];
    const float* Wa   = (const float*)d_in[4];
    const float* Va   = (const float*)d_in[5];
    const float* kern = (const float*)d_in[6];
    const float* rec  = (const float*)d_in[7];
    const float* bias = (const float*)d_in[8];
    float* out = (float*)d_out;

    char* ws = (char*)d_ws;
    size_t off = 0;
    auto alloc = [&](size_t bytes) -> void* {
        off = (off + 255) & ~(size_t)255;
        void* p = ws + off;
        off += bytes;
        return p;
    };

    // Per-step L2 working set = uxpbT (32MB) + x_bf16 (64MB) = 96MB < 192MB L2.
    __bf16* xb16  = (__bf16*)alloc((size_t)B_ * T_ * D_ * 2);      // 64 MB
    __bf16* uxpbT = (__bf16*)alloc((size_t)B_ * U_ * T_ * 2);      // 32 MB, [b][u][t]
    __bf16* UaP   = (__bf16*)alloc((size_t)D_ * U_ * 2);           // packed B
    __bf16* kernP = (__bf16*)alloc((size_t)D_ * 4 * U_ * 2);       // packed B
    __bf16* recP  = (__bf16*)alloc((size_t)U_ * 4 * U_ * 2);       // packed B
    float*  h0    = (float*) alloc((size_t)B_ * U_ * 4);
    float*  h1    = (float*) alloc((size_t)B_ * U_ * 4);
    float*  c     = (float*) alloc((size_t)B_ * U_ * 4);
    float*  ctx   = (float*) alloc((size_t)B_ * D_ * 4);
    float*  e     = (float*) alloc((size_t)B_ * T_ * 4);
    (void)ws_size; (void)in_sizes; (void)n_in; (void)out_size;

    // ---- one-time setup ----
    int n4 = (B_ * T_ * D_) / 4;
    cvt_x<<<n4 / 256, 256, 0, stream>>>(x, xb16, n4);
    pack_b<<<(U_ / 16) * (D_ / 32) * 32 / 256, 256, 0, stream>>>(Ua, UaP, D_, U_);
    pack_b<<<(4 * U_ / 16) * (D_ / 32) * 32 / 256, 256, 0, stream>>>(kern, kernP, D_, 4 * U_);
    pack_b<<<(4 * U_ / 16) * (U_ / 32) * 32 / 256, 256, 0, stream>>>(rec, recP, U_, 4 * U_);
    uxpb_gemm<<<(B_ * T_ / 16) * (U_ / 16) / 8, 256, 0, stream>>>(xb16, UaP, ba, uxpbT);
    init_state<<<B_, U_, 0, stream>>>(x, Ws, h0, c);

    // ---- 256 sequential decode steps (h f32 ping-pong) ----
    for (int step = 0; step < TDEC_; ++step) {
        float* hin  = (step & 1) ? h1 : h0;
        float* hout = (step & 1) ? h0 : h1;
        attn_scores<<<B_ * 4, 256, 0, stream>>>(uxpbT, hin, Wa, Va, e, ctx);
        softmax_ctx<<<B_ * 8, 256, 0, stream>>>(e, xb16, ctx);
        lstm_gates<<<32, 32, 0, stream>>>(ctx, kernP, hin, recP, bias, hout, c, out, step);
    }
}
